// VannilaSlot_51780125720773
// MI455X (gfx1250) — matrix operand order repaired and checked by reading, not compiled, not run on hardware
//
#include <hip/hip_runtime.h>
#include <hip/hip_bf16.h>
#include <cstdint>
#include <cstddef>

typedef __bf16 bf16_t;
typedef __attribute__((ext_vector_type(16))) __bf16 v16bf;
typedef __attribute__((ext_vector_type(8)))  __bf16 v8bf;
typedef __attribute__((ext_vector_type(8)))  float  v8f;
typedef __attribute__((ext_vector_type(4)))  float  v4f;
typedef __attribute__((ext_vector_type(2)))  float  v2f;

// ---- problem constants ----
#define BATCH    64
#define NKV      8192
#define DIM      64
#define NSLOTS   8
#define HEADS    4
#define C32      32          // HEADS*NSLOTS, joint-softmax width
#define MLPHID   128
#define EPSA     1e-8f

// ---- attn kernel tiling ----
#define AWAVES   4
#define ATHREADS 128
#define ANT      8                       // 16-row tiles per wave
#define ROWS_IT  (AWAVES * 16)           // 64 rows per stage iteration
#define ROWS_BLK (ROWS_IT * ANT)         // 512 rows per block
#define NCHUNKS  (NKV / ROWS_BLK)        // 16

// ---------------- CDNA5 async global->LDS copy helpers ----------------
// Param 1 of the builtin is (per clang diagnostic):
//   __attribute__((__vector_size__(4*sizeof(int)))) int __device__ *
// i.e. a global-AS pointer to a 4 x i32 vector. LDS side mirrors in AS3.
typedef int vsi4 __attribute__((__vector_size__(4 * sizeof(int))));
typedef __attribute__((address_space(1))) vsi4* gptr_v4i;
typedef __attribute__((address_space(3))) vsi4* lptr_v4i;

#if defined(__has_builtin)
#if __has_builtin(__builtin_amdgcn_global_load_async_to_lds_b128)
#define ASYNC_VIA_BUILTIN 1
#endif
#endif

// Copy 16 bytes global -> LDS, tracked by ASYNCcnt.
__device__ __forceinline__ void async_cp16(const void* g, void* l) {
#if defined(ASYNC_VIA_BUILTIN)
  __builtin_amdgcn_global_load_async_to_lds_b128((gptr_v4i)g, (lptr_v4i)l, 0,
                                                 0);
#else
  unsigned lds_off = (unsigned)(__SIZE_TYPE__)(lptr_v4i)l;
  unsigned long long ga = (unsigned long long)(__SIZE_TYPE__)g;
  asm volatile("global_load_async_to_lds_b128 %0, %1, off"
               :
               : "v"(lds_off), "v"(ga)
               : "memory");
#endif
}

__device__ __forceinline__ void wait_async(int pending) {
#if defined(__has_builtin)
#if __has_builtin(__builtin_amdgcn_s_wait_asynccnt)
  if (pending)
    __builtin_amdgcn_s_wait_asynccnt(8);
  else
    __builtin_amdgcn_s_wait_asynccnt(0);
  asm volatile("" ::: "memory");
  return;
#endif
#endif
  if (pending)
    asm volatile("s_wait_asynccnt 8" ::: "memory");
  else
    asm volatile("s_wait_asynccnt 0" ::: "memory");
}

__device__ __forceinline__ v8f wmma_bf16(v16bf a, v16bf b, v8f c) {
  return __builtin_amdgcn_wmma_f32_16x16x32_bf16(false, a, false, b, (short)0, c,
                                                 false, false);
}

// Build a 16x32 bf16 A fragment from a row-major [16][64] bf16 LDS tile.
// lane: M = lane&15, kbase = 32*s + 8*(lane>>4); elems 0..7 -> K=kbase+e,
// elems 8..15 -> K=kbase+16+(e-8)  (ISA 7.12.2, 16-bit A matrix).
__device__ __forceinline__ v16bf load_afrag16(const __bf16* row, int kb) {
  v8bf lo = *(const v8bf*)(row + kb);
  v8bf hi = *(const v8bf*)(row + kb + 16);
  v16bf r;
#pragma unroll
  for (int e = 0; e < 8; ++e) { r[e] = lo[e]; r[e + 8] = hi[e]; }
  return r;
}

// =====================================================================
// Kernel Q: per-batch slot LN + q projection into masked Qmat[64][32]
// (block-diagonal per head), plus zeroing of colsum/upd accumulators.
// =====================================================================
__global__ __launch_bounds__(64) void slotq_kernel(
    const float* __restrict__ slots_src, const float* __restrict__ lng,
    const float* __restrict__ lnb, const float* __restrict__ Wq,
    bf16_t* __restrict__ qmat, float* __restrict__ colsum,
    float* __restrict__ upd) {
  int b = blockIdx.x;
  int t = threadIdx.x;  // 64 threads
  __shared__ float s_lds[NSLOTS][DIM];
  __shared__ float q_lds[NSLOTS][DIM];

  if (t < NSLOTS) {
    const float* row = slots_src + (size_t)(b * NSLOTS + t) * DIM;
    float m = 0.f;
    for (int j = 0; j < DIM; ++j) m += row[j];
    m *= (1.f / DIM);
    float v = 0.f;
    for (int j = 0; j < DIM; ++j) { float d = row[j] - m; v += d * d; }
    v *= (1.f / DIM);
    float rs = rsqrtf(v + 1e-5f);
    for (int j = 0; j < DIM; ++j)
      s_lds[t][j] = (row[j] - m) * rs * lng[j] + lnb[j];
  }
  __syncthreads();

  // thread t computes q[:, t]
  for (int qi = 0; qi < NSLOTS; ++qi) {
    float acc = 0.f;
    for (int d = 0; d < DIM; ++d) acc += s_lds[qi][d] * Wq[d * DIM + t];
    q_lds[qi][t] = acc;
  }
  __syncthreads();

  // masked Qmat: Qmat[k = h*16+dh][c = h*8+qi] = q[qi][k], zero elsewhere
  bf16_t* qm = qmat + (size_t)b * DIM * C32;
  int h = t >> 4;
  for (int c = 0; c < C32; ++c) qm[t * C32 + c] = (bf16_t)0.f;
  for (int qi = 0; qi < NSLOTS; ++qi)
    qm[t * C32 + (h * NSLOTS + qi)] = (bf16_t)q_lds[qi][t];

  // zero per-iteration accumulators for this batch
  if (t < C32) colsum[b * C32 + t] = 0.f;
  float* u = upd + (size_t)b * C32 * DIM;
  for (int i = t; i < C32 * DIM; i += 64) u[i] = 0.f;
}

// =====================================================================
// Fused attention kernel: async double-buffered input stream ->
// LN -> K/V proj (bf16 WMMA) -> logits (bf16 WMMA vs masked Q) ->
// joint softmax(32) -> attn_vis, colsum atomics, attn^T @ v partials
// (f32 WMMA 16x16x4).
// =====================================================================
__global__ __launch_bounds__(ATHREADS) void attn_kernel(
    const float* __restrict__ inputs, const float* __restrict__ lnin_g,
    const float* __restrict__ lnin_b, const float* __restrict__ Wk,
    const float* __restrict__ Wv, const bf16_t* __restrict__ qmat,
    float* __restrict__ colsum, float* __restrict__ upd,
    float* __restrict__ vcolsum, float* __restrict__ attn_vis,
    int first_iter, int last_iter) {
  const int b = blockIdx.x / NCHUNKS;
  const int chunk = blockIdx.x % NCHUNKS;
  const int n0 = chunk * ROWS_BLK;
  const int t = threadIdx.x;
  const int wave = t >> 5;
  const int lane = t & 31;
  const int lhalf = lane >> 4;
  const int l16 = lane & 15;

  // Dynamic LDS: double-buffered raw f32 input tiles [2][ROWS_IT][DIM].
  extern __shared__ __align__(16) float xraw[];

  // Each thread owns 128B of the tile: row = t>>1, cols (t&1)*32..+31.
  const int my_row = t >> 1;
  const int my_half = t & 1;

  // Issue async loads for one tile into buffer `buf`.
  auto issue_tile = [&](int buf, int rowbase) {
    const float* g =
        inputs + ((size_t)b * NKV + rowbase + my_row) * DIM + my_half * 32;
    float* l = xraw + (size_t)buf * ROWS_IT * DIM + my_row * DIM + my_half * 32;
#pragma unroll
    for (int q = 0; q < 8; ++q) async_cp16(g + q * 4, l + q * 4);
  };

  // B operands pre-permuted into fragment order: one aligned v16bf per lane.
  __shared__ __align__(32) __bf16 wkB[2][4][32][16];
  __shared__ __align__(32) __bf16 wvB[2][4][32][16];
  __shared__ __align__(32) __bf16 qB[2][2][32][16];
  __shared__ float lng[DIM], lnb[DIM];
  __shared__ __align__(16) __bf16 xA[AWAVES][16][DIM];   // LN'd input tiles
  __shared__ __align__(16) __bf16 kA[AWAVES][16][DIM];   // k tiles (bf16)
  __shared__ __align__(16) __bf16 vT[AWAVES][16][DIM];   // v tiles (bf16)
  __shared__ __align__(16) float at[AWAVES][16][C32];    // logits -> attn
  __shared__ float upd_red[C32][DIM];
  __shared__ float cs_red[C32];
  __shared__ float vs_red[DIM];

  // Kick off the first tile before doing the LDS setup work.
  issue_tile(0, n0);

  // ---- stage 0: load LN params, permute Wk/Wv/Qmat into B-frag layout ----
  for (int i = t; i < DIM; i += ATHREADS) { lng[i] = lnin_g[i]; lnb[i] = lnin_b[i]; }
  for (int i = t; i < 2 * 4 * 32 * 16; i += ATHREADS) {
    int e = i & 15, L = (i >> 4) & 31, tn = (i >> 9) & 3, s = (i >> 11) & 1;
    int N = tn * 16 + (L & 15);
    int K = s * 32 + (L >> 4) * 8 + (e & 7) + (e >> 3) * 16;  // B layout: table K order (matches A)
    wkB[s][tn][L][e] = (__bf16)(0.25f * Wk[K * DIM + N]);  // fold 1/sqrt(16)
    wvB[s][tn][L][e] = (__bf16)(Wv[K * DIM + N]);
  }
  const bf16_t* qm = qmat + (size_t)b * DIM * C32;
  for (int i = t; i < 2 * 2 * 32 * 16; i += ATHREADS) {
    int e = i & 15, L = (i >> 4) & 31, tn = (i >> 9) & 1, s = (i >> 10) & 1;
    int N = tn * 16 + (L & 15);
    int K = s * 32 + (L >> 4) * 8 + (e & 7) + (e >> 3) * 16;
    qB[s][tn][L][e] = qm[K * C32 + N];
  }
  {
    float* ur = &upd_red[0][0];
    for (int i = t; i < C32 * DIM; i += ATHREADS) ur[i] = 0.f;
    if (t < C32) cs_red[t] = 0.f;
    if (t < DIM) vs_red[t] = 0.f;
  }
  __syncthreads();

  v8f uacc[2][4];
#pragma unroll
  for (int mt = 0; mt < 2; ++mt)
#pragma unroll
    for (int tn = 0; tn < 4; ++tn)
#pragma unroll
      for (int r = 0; r < 8; ++r) uacc[mt][tn][r] = 0.f;
  float csacc = 0.f, vsacc0 = 0.f, vsacc1 = 0.f;

  for (int it = 0; it < ANT; ++it) {
    const int rowbase = n0 + it * ROWS_IT;

    // ---- stage 1: prefetch next tile, wait current, LayerNorm -> xA ----
    {
      if (it + 1 < ANT) {
        issue_tile((it + 1) & 1, rowbase + ROWS_IT);
        wait_async(1);  // loads complete in order: current tile is resident
      } else {
        wait_async(0);
      }
      const v4f* srcv = (const v4f*)(xraw + (size_t)(it & 1) * ROWS_IT * DIM +
                                     my_row * DIM + my_half * 32);
      float x[32];
      float s1 = 0.f;
#pragma unroll
      for (int q = 0; q < 8; ++q) {
        v4f xv = srcv[q];
#pragma unroll
        for (int j = 0; j < 4; ++j) { x[q * 4 + j] = xv[j]; s1 += xv[j]; }
      }
      float mean = (s1 + __shfl_xor(s1, 1)) * (1.f / DIM);
      float s2 = 0.f;
#pragma unroll
      for (int j = 0; j < 32; ++j) { float d = x[j] - mean; s2 += d * d; }
      float var = (s2 + __shfl_xor(s2, 1)) * (1.f / DIM);
      float rs = rsqrtf(var + 1e-5f);
      int wv_ = my_row >> 4, rr = my_row & 15;
#pragma unroll
      for (int j = 0; j < 32; ++j) {
        int col = my_half * 32 + j;
        xA[wv_][rr][col] = (__bf16)((x[j] - mean) * rs * lng[col] + lnb[col]);
      }
    }
    __syncthreads();

    // ---- stage 2: K/V projection, 16x64 per wave (bf16 WMMA) ----
    {
      v16bf ax[2];
#pragma unroll
      for (int s = 0; s < 2; ++s)
        ax[s] = load_afrag16(&xA[wave][l16][0], s * 32 + lhalf * 8);

      v8f kacc[4], vacc[4];
#pragma unroll
      for (int tn = 0; tn < 4; ++tn)
#pragma unroll
        for (int r = 0; r < 8; ++r) { kacc[tn][r] = 0.f; vacc[tn][r] = 0.f; }
#pragma unroll
      for (int s = 0; s < 2; ++s)
#pragma unroll
        for (int tn = 0; tn < 4; ++tn) {
          v16bf bk = *(const v16bf*)&wkB[s][tn][lane][0];
          v16bf bv = *(const v16bf*)&wvB[s][tn][lane][0];
          kacc[tn] = wmma_bf16(ax[s], bk, kacc[tn]);
          vacc[tn] = wmma_bf16(ax[s], bv, vacc[tn]);
        }
      // scatter C-layout results to row-major LDS (bf16)
#pragma unroll
      for (int tn = 0; tn < 4; ++tn)
#pragma unroll
        for (int r = 0; r < 8; ++r) {
          int m = r + lhalf * 8;
          int col = tn * 16 + l16;
          kA[wave][m][col] = (__bf16)kacc[tn][r];
          vT[wave][m][col] = (__bf16)vacc[tn][r];
        }
    }
    __syncthreads();

    // ---- stage 3: logits = k(16x64) @ Qmat(64x32) ----
    {
      v16bf ka[2];
#pragma unroll
      for (int s = 0; s < 2; ++s)
        ka[s] = load_afrag16(&kA[wave][l16][0], s * 32 + lhalf * 8);
      v8f lacc[2];
#pragma unroll
      for (int tn = 0; tn < 2; ++tn)
#pragma unroll
        for (int r = 0; r < 8; ++r) lacc[tn][r] = 0.f;
#pragma unroll
      for (int s = 0; s < 2; ++s)
#pragma unroll
        for (int tn = 0; tn < 2; ++tn) {
          v16bf bq = *(const v16bf*)&qB[s][tn][lane][0];
          lacc[tn] = wmma_bf16(ka[s], bq, lacc[tn]);
        }
#pragma unroll
      for (int tn = 0; tn < 2; ++tn)
#pragma unroll
        for (int r = 0; r < 8; ++r)
          at[wave][r + lhalf * 8][tn * 16 + l16] = lacc[tn][r];
    }
    __syncthreads();

    // ---- stage 4: joint softmax over 32 (2 lanes per row) ----
    {
      float lg[16];
      const v4f* arow = (const v4f*)&at[wave][l16][lhalf * 16];
#pragma unroll
      for (int q = 0; q < 4; ++q) {
        v4f v = arow[q];
#pragma unroll
        for (int j = 0; j < 4; ++j) lg[q * 4 + j] = v[j];
      }
      float mx = lg[0];
#pragma unroll
      for (int j = 1; j < 16; ++j) mx = fmaxf(mx, lg[j]);
      mx = fmaxf(mx, __shfl_xor(mx, 16));
      float sum = 0.f;
#pragma unroll
      for (int j = 0; j < 16; ++j) { lg[j] = __expf(lg[j] - mx); sum += lg[j]; }
      sum += __shfl_xor(sum, 16);
      float inv = 1.f / sum;
#pragma unroll
      for (int j = 0; j < 16; ++j) lg[j] *= inv;

      if (last_iter) {  // attn_vis = sum over heads
        float pv[8];
#pragma unroll
        for (int qi = 0; qi < 8; ++qi) pv[qi] = lg[qi] + lg[qi + 8];
#pragma unroll
        for (int qi = 0; qi < 8; ++qi) pv[qi] += __shfl_xor(pv[qi], 16);
        int n = rowbase + wave * 16 + l16;
        v4f out;
#pragma unroll
        for (int j = 0; j < 4; ++j) out[j] = pv[lhalf * 4 + j];
        *(v4f*)&attn_vis[((size_t)b * NKV + n) * NSLOTS + lhalf * 4] = out;
      }
      v4f* awr = (v4f*)&at[wave][l16][lhalf * 16];
#pragma unroll
      for (int q = 0; q < 4; ++q) {
        v4f v;
#pragma unroll
        for (int j = 0; j < 4; ++j) v[j] = lg[q * 4 + j];
        awr[q] = v;
      }
    }
    __syncthreads();

    // ---- stage 5: column sums + attn^T @ v partials (f32 WMMA K=4) ----
#pragma unroll
    for (int r = 0; r < 16; ++r) csacc += at[wave][r][lane];
    if (first_iter) {
#pragma unroll
      for (int r = 0; r < 16; ++r) {
        vsacc0 += (float)vT[wave][r][lane];
        vsacc1 += (float)vT[wave][r][lane + 32];
      }
    }
#pragma unroll
    for (int s = 0; s < 4; ++s) {
      int k0 = s * 4 + lhalf * 2;
#if __has_builtin(__builtin_amdgcn_wmma_f32_16x16x4_f32)
      v2f bfrag[4];
#pragma unroll
      for (int tn = 0; tn < 4; ++tn) {
        bfrag[tn][0] = (float)vT[wave][k0][tn * 16 + l16];
        bfrag[tn][1] = (float)vT[wave][k0 + 1][tn * 16 + l16];
      }
#pragma unroll
      for (int mt = 0; mt < 2; ++mt) {
        v2f af;
        af[0] = at[wave][k0][mt * 16 + l16];
        af[1] = at[wave][k0 + 1][mt * 16 + l16];
#pragma unroll
        for (int tn = 0; tn < 4; ++tn)
          uacc[mt][tn] = __builtin_amdgcn_wmma_f32_16x16x4_f32(
              false, af, false, bfrag[tn], (short)0, uacc[mt][tn], false, false);
      }
#else
#pragma unroll
      for (int mt = 0; mt < 2; ++mt)
#pragma unroll
        for (int tn = 0; tn < 4; ++tn)
#pragma unroll
          for (int r = 0; r < 8; ++r) {
            int m = mt * 16 + r + lhalf * 8;
            int n = tn * 16 + l16;
            float a = 0.f;
            for (int kk = 0; kk < 4; ++kk)
              a += at[wave][s * 4 + kk][m] * (float)vT[wave][s * 4 + kk][n];
            uacc[mt][tn][r] += a;
          }
#endif
    }
    __syncthreads();
  }

  // ---- stage 6: block-level LDS reduction then global atomics ----
#pragma unroll
  for (int mt = 0; mt < 2; ++mt)
#pragma unroll
    for (int tn = 0; tn < 4; ++tn)
#pragma unroll
      for (int r = 0; r < 8; ++r) {
        int c = mt * 16 + r + lhalf * 8;
        int col = tn * 16 + l16;
        atomicAdd(&upd_red[c][col], uacc[mt][tn][r]);
      }
  atomicAdd(&cs_red[lane], csacc);
  if (first_iter) {
    atomicAdd(&vs_red[lane], vsacc0);
    atomicAdd(&vs_red[lane + 32], vsacc1);
  }
  __syncthreads();
  float* ug = upd + (size_t)b * C32 * DIM;
  const float* ur = &upd_red[0][0];
  for (int i = t; i < C32 * DIM; i += ATHREADS) atomicAdd(&ug[i], ur[i]);
  if (t < C32) atomicAdd(&colsum[b * C32 + t], cs_red[t]);
  if (first_iter && t < DIM) atomicAdd(&vcolsum[b * DIM + t], vs_red[t]);
}

// =====================================================================
// Kernel G: normalize updates (with deferred EPS), GRU cell, MLP.
// =====================================================================
__device__ __forceinline__ float sigm(float x) { return 1.f / (1.f + __expf(-x)); }

__global__ __launch_bounds__(64) void gru_mlp_kernel(
    const float* __restrict__ upd_raw, const float* __restrict__ colsum,
    const float* __restrict__ vcolsum, const float* __restrict__ slots_in,
    const float* __restrict__ W_ih, const float* __restrict__ W_hh,
    const float* __restrict__ b_ih, const float* __restrict__ b_hh,
    const float* __restrict__ lnm_g, const float* __restrict__ lnm_b,
    const float* __restrict__ mlp_W1, const float* __restrict__ mlp_b1,
    const float* __restrict__ mlp_W2, const float* __restrict__ mlp_b2,
    float* __restrict__ slots_out, float* __restrict__ dout_slots, int last) {
  int b = blockIdx.x;
  int t = threadIdx.x;  // 64
  __shared__ float s_upd[NSLOTS][DIM], s_h[NSLOTS][DIM];
  __shared__ float s_gx[NSLOTS][3 * DIM], s_gh[NSLOTS][3 * DIM];
  __shared__ float s_new[NSLOTS][DIM], s_ln[NSLOTS][DIM];
  __shared__ float s_m[NSLOTS][MLPHID];

  const float* U = upd_raw + (size_t)b * C32 * DIM;
  const float* CS = colsum + b * C32;
  const float* VS = vcolsum + b * DIM;
  for (int idx = t; idx < NSLOTS * DIM; idx += 64) {
    int qi = idx >> 6, j = idx & 63, h = j >> 4;
    int c = h * NSLOTS + qi;
    float num = U[c * DIM + j] + EPSA * VS[j];
    float den = CS[c] + (float)NKV * EPSA;
    s_upd[qi][j] = num / den;
    s_h[qi][j] = slots_in[(size_t)(b * NSLOTS + qi) * DIM + j];
  }
  __syncthreads();

  for (int idx = t; idx < NSLOTS * 3 * DIM; idx += 64) {
    int qi = idx / (3 * DIM), g = idx % (3 * DIM);
    float ax = b_ih[g], ah = b_hh[g];
    for (int d = 0; d < DIM; ++d) {
      ax += s_upd[qi][d] * W_ih[g * DIM + d];
      ah += s_h[qi][d] * W_hh[g * DIM + d];
    }
    s_gx[qi][g] = ax;
    s_gh[qi][g] = ah;
  }
  __syncthreads();

  for (int idx = t; idx < NSLOTS * DIM; idx += 64) {
    int qi = idx >> 6, j = idx & 63;
    float r = sigm(s_gx[qi][j] + s_gh[qi][j]);
    float z = sigm(s_gx[qi][DIM + j] + s_gh[qi][DIM + j]);
    float n = tanhf(s_gx[qi][2 * DIM + j] + r * s_gh[qi][2 * DIM + j]);
    s_new[qi][j] = (1.f - z) * n + z * s_h[qi][j];
  }
  __syncthreads();

  if (t < NSLOTS) {
    float m = 0.f;
    for (int j = 0; j < DIM; ++j) m += s_new[t][j];
    m *= (1.f / DIM);
    float v = 0.f;
    for (int j = 0; j < DIM; ++j) { float d = s_new[t][j] - m; v += d * d; }
    v *= (1.f / DIM);
    float rs = rsqrtf(v + 1e-5f);
    for (int j = 0; j < DIM; ++j)
      s_ln[t][j] = (s_new[t][j] - m) * rs * lnm_g[j] + lnm_b[j];
  }
  __syncthreads();

  for (int idx = t; idx < NSLOTS * MLPHID; idx += 64) {
    int qi = idx >> 7, mc = idx & 127;
    float a = mlp_b1[mc];
    for (int d = 0; d < DIM; ++d) a += s_ln[qi][d] * mlp_W1[d * MLPHID + mc];
    s_m[qi][mc] = fmaxf(a, 0.f);
  }
  __syncthreads();

  for (int idx = t; idx < NSLOTS * DIM; idx += 64) {
    int qi = idx >> 6, j = idx & 63;
    float a = mlp_b2[j];
    for (int m = 0; m < MLPHID; ++m) a += s_m[qi][m] * mlp_W2[m * DIM + j];
    float val = s_new[qi][j] + a;
    slots_out[(size_t)(b * NSLOTS + qi) * DIM + j] = val;
    if (last) dout_slots[(size_t)(b * NSLOTS + qi) * DIM + j] = val;
  }
}

// =====================================================================
extern "C" void kernel_launch(void* const* d_in, const int* in_sizes, int n_in,
                              void* d_out, int out_size, void* d_ws,
                              size_t ws_size, hipStream_t stream) {
  const float* inputs   = (const float*)d_in[0];
  const float* slots0   = (const float*)d_in[1];
  const float* ln_in_g  = (const float*)d_in[2];
  const float* ln_in_b  = (const float*)d_in[3];
  const float* ln_sl_g  = (const float*)d_in[4];
  const float* ln_sl_b  = (const float*)d_in[5];
  const float* ln_ml_g  = (const float*)d_in[6];
  const float* ln_ml_b  = (const float*)d_in[7];
  const float* Wq       = (const float*)d_in[8];
  const float* Wk       = (const float*)d_in[9];
  const float* Wv       = (const float*)d_in[10];
  const float* W_ih     = (const float*)d_in[11];
  const float* W_hh     = (const float*)d_in[12];
  const float* b_ih     = (const float*)d_in[13];
  const float* b_hh     = (const float*)d_in[14];
  const float* mlp_W1   = (const float*)d_in[15];
  const float* mlp_b1   = (const float*)d_in[16];
  const float* mlp_W2   = (const float*)d_in[17];
  const float* mlp_b2   = (const float*)d_in[18];

  char* ws = (char*)d_ws;
  size_t off = 0;
  bf16_t* qmat = (bf16_t*)(ws + off); off += (size_t)BATCH * DIM * C32 * 2;   // 256 KB
  float* colsum = (float*)(ws + off); off += (size_t)BATCH * C32 * 4;         // 8 KB
  float* upd = (float*)(ws + off);    off += (size_t)BATCH * C32 * DIM * 4;   // 512 KB
  float* vcolsum = (float*)(ws + off); off += (size_t)BATCH * DIM * 4;        // 16 KB
  float* slots_buf = (float*)(ws + off); off += (size_t)BATCH * NSLOTS * DIM * 4;

  float* out_slots = (float*)d_out;
  float* attn_vis = (float*)d_out + (size_t)BATCH * NSLOTS * DIM;

  (void)hipMemsetAsync(vcolsum, 0, (size_t)BATCH * DIM * 4, stream);
  (void)hipMemcpyAsync(slots_buf, slots0, (size_t)BATCH * NSLOTS * DIM * 4,
                       hipMemcpyDeviceToDevice, stream);

  const size_t dyn_lds = (size_t)2 * ROWS_IT * DIM * sizeof(float);  // 32 KB

  for (int it = 0; it < 3; ++it) {
    slotq_kernel<<<BATCH, 64, 0, stream>>>(slots_buf, ln_sl_g, ln_sl_b, Wq,
                                           qmat, colsum, upd);
    attn_kernel<<<BATCH * NCHUNKS, ATHREADS, dyn_lds, stream>>>(
        inputs, ln_in_g, ln_in_b, Wk, Wv, qmat, colsum, upd, vcolsum, attn_vis,
        it == 0 ? 1 : 0, it == 2 ? 1 : 0);
    gru_mlp_kernel<<<BATCH, 64, 0, stream>>>(
        upd, colsum, vcolsum, slots_buf, W_ih, W_hh, b_ih, b_hh, ln_ml_g,
        ln_ml_b, mlp_W1, mlp_b1, mlp_W2, mlp_b2, slots_buf, out_slots,
        it == 2 ? 1 : 0);
  }
}
